// Network_81406810129159
// MI455X (gfx1250) — matrix operand order repaired and checked by reading, not compiled, not run on hardware
//
#include <hip/hip_runtime.h>
#include <hip/hip_bf16.h>
#include <stdint.h>
#include <stddef.h>

// ---------------------------------------------------------------------------
// Equivariant GNN forward for MI455X (gfx1250, wave32).
// Per-edge radial MLPs (the FLOP-dominant part, ~2.3e10 FLOPs) run on
// v_wmma_f32_16x16x32_bf16, fused with gather / message / atomic-scatter so
// per-edge intermediates never touch HBM. Everything else is bandwidth-bound
// helper kernels. The reference's v2_1e branch is dead code and is skipped.
// Workspace use: ~140 MB.
// ---------------------------------------------------------------------------

#define NNODES  30000
#define NEDGES  500000
#define NGRAPHS 64
#define NTILES  (NEDGES / 16)   // 31250, exact
#define EW      4               // waves per block in edge kernels

typedef __attribute__((ext_vector_type(16))) __bf16          v16bf;
typedef __attribute__((ext_vector_type(8)))  float           v8f;
typedef __attribute__((ext_vector_type(16))) unsigned short  v16us;
typedef __attribute__((ext_vector_type(8)))  unsigned short  v8us;

constexpr float PI_F        = 3.14159265358979323846f;
constexpr float R_MAX_F     = 5.0f;
constexpr float SQRT3_F     = 1.7320508075688772f;
constexpr float EMB_SCALE   = 3.1622776601683795f / 1.12f;   // sqrt(10)/1.12
constexpr float INV_SQRT_NN = 0.22360679774997896f;          // 1/sqrt(20)
constexpr float ISQ3_F      = 0.57735026918962576f;
constexpr float CS_F        = 0.3826834323650898f;           // sin(pi/8)
constexpr float CX_F        = 0.9238795325112867f;           // cos(pi/8)
constexpr float IS16        = 0.25f;                          // 1/sqrt(16)
constexpr float IS32        = 0.17677669529663687f;           // 1/sqrt(32)
constexpr float IS64        = 0.125f;                         // 1/sqrt(64)
constexpr float IS10        = 0.31622776601683794f;           // 1/sqrt(10)
constexpr float IS100       = 0.1f;                           // 1/sqrt(100)

__device__ __forceinline__ unsigned short f2bf(float f) {
  unsigned u = __builtin_bit_cast(unsigned, f);
  u += 0x7FFFu + ((u >> 16) & 1u);          // round-to-nearest-even
  return (unsigned short)(u >> 16);
}
__device__ __forceinline__ float siluf(float x) { return x / (1.0f + __expf(-x)); }
__device__ __forceinline__ float sigmf(float x) { return 1.0f / (1.0f + __expf(-x)); }

// ---------------------------------------------------------------------------
// Per-wave radial MLP on WMMA: 16 edges -> NT*16 radial channels (f32, D layout)
//   GEMM1: A = emb[16x32] (10 valid K), B = w1 frags (7 N-tiles) -> H[16x112]
//   silu, transpose via per-wave LDS slice [16 rows x 128 cols, bf16]
//   GEMM2: A = H[16x128], B = w2 frags (4 K-chunks x NT tiles)
// A-fragment layout used here (16-bit, 16x32): lane<16 holds M=lane, K 0..15;
// lane>=16 holds M=lane-16, K 16..31.
// B-fragment layout (32x16): lane holds column n=lane%16, K (lane<16?0..15:16..31).
// D layout: lane holds column n=lane&15, rows r + 8*(lane>=16).
// ---------------------------------------------------------------------------
template<int NT>
__device__ __forceinline__ void radial16(int lane, int e0,
    const unsigned short* __restrict__ emb16,
    const unsigned short* __restrict__ w1f,
    const unsigned short* __restrict__ w2f,
    unsigned short* __restrict__ lds, v8f* acc)
{
  const int lm = lane & 15;
  const int hi = lane >> 4;

  // A fragment from packed emb (16 bf16 per edge, entries 10..15 are zero).
  const size_t e = (size_t)(e0 + lm);
  v16us au = {};
  if (hi == 0) au = *(const v16us*)(emb16 + e * 16);   // hi lanes hold K 16..31: all 0 (K>=10 padded)
  v16bf a = __builtin_bit_cast(v16bf, au);

  // GEMM1: 7 output tiles (N = 112, cols 100..111 give silu(0)=0).
  v8f h[7];
  #pragma unroll
  for (int t = 0; t < 7; ++t) {
    v16bf b = __builtin_bit_cast(v16bf, *(const v16us*)(w1f + (size_t)(t * 32 + lane) * 16));
    v8f c = {};
    h[t] = __builtin_amdgcn_wmma_f32_16x16x32_bf16(false, a, false, b, (short)0, c, false, false);
  }

  // Zero-pad LDS columns 112..127 (never written by silu stores).
  {
    const int col = 112 + hi * 8;
    #pragma unroll
    for (int i = 0; i < 8; ++i) lds[lm * 128 + col + i] = 0;
  }
  // silu + transpose: lane holds column 16t+lm, rows r + 8*hi.
  #pragma unroll
  for (int t = 0; t < 7; ++t) {
    #pragma unroll
    for (int r = 0; r < 8; ++r)
      lds[(r + hi * 8) * 128 + t * 16 + lm] = f2bf(siluf(h[t][r]));
  }

  #pragma unroll
  for (int t = 0; t < NT; ++t) acc[t] = {};

  // GEMM2: K = 128 as 4 chunks of 32; A rebuilt from LDS per chunk.
  #pragma unroll
  for (int c = 0; c < 4; ++c) {
    v8us p0 = *(const v8us*)(lds + lm * 128 + c * 32 + hi * 16);
    v8us p1 = *(const v8us*)(lds + lm * 128 + c * 32 + hi * 16 + 8);
    v16us av;
    #pragma unroll
    for (int i = 0; i < 8; ++i) { av[i] = p0[i]; av[8 + i] = p1[i]; }
    v16bf a2 = __builtin_bit_cast(v16bf, av);
    #pragma unroll
    for (int t = 0; t < NT; ++t) {
      v16bf b = __builtin_bit_cast(v16bf,
          *(const v16us*)(w2f + (size_t)((t * 4 + c) * 32 + lane) * 16));
      acc[t] = __builtin_amdgcn_wmma_f32_16x16x32_bf16(false, a2, false, b, (short)0, acc[t], false, false);
    }
  }
}

// ---------------------------------------------------------------------------
// Edge geometry: d, cutoff a0, vector basis a1 = cut*sqrt(3)*unit, bf16 emb.
// ---------------------------------------------------------------------------
__global__ void k_edge_geom(const float* __restrict__ evec,
                            float* __restrict__ a0g, float* __restrict__ a1g,
                            unsigned short* __restrict__ emb16, int ne)
{
  int e = blockIdx.x * blockDim.x + threadIdx.x;
  if (e >= ne) return;
  float vx = evec[(size_t)e * 3], vy = evec[(size_t)e * 3 + 1], vz = evec[(size_t)e * 3 + 2];
  float d   = sqrtf(vx * vx + vy * vy + vz * vz);
  float inv = 1.0f / d;
  float u   = 2.0f * (d / R_MAX_F - 1.0f);
  float cut = (u > 0.0f) ? 0.0f : ((u < -1.0f) ? 1.0f : 0.5f * (1.0f - cosf(PI_F * u)));
  a0g[e] = cut;
  float c3 = cut * SQRT3_F * inv;
  a1g[(size_t)e * 3]     = c3 * vx;
  a1g[(size_t)e * 3 + 1] = c3 * vy;
  a1g[(size_t)e * 3 + 2] = c3 * vz;
  const float step = R_MAX_F / 9.0f;
  #pragma unroll
  for (int b = 0; b < 16; ++b) {
    unsigned short hbits = 0;
    if (b < 10) {
      float diff = (d - step * (float)b) / step;
      hbits = f2bf(__expf(-diff * diff) * EMB_SCALE);
    }
    emb16[(size_t)e * 16 + b] = hbits;
  }
}

// Pack a KxN f32 weight (row-major) into bf16 WMMA-B fragments.
// out[((t*kT + c)*32 + lane)*16 + v] = W[32c + (lane<16?0:16) + v][16t + (lane&15)] * scale
__global__ void k_pack_b(const float* __restrict__ W, int K, int N, int kT, int nT,
                         float scale, unsigned short* __restrict__ out)
{
  int i = blockIdx.x * blockDim.x + threadIdx.x;
  int total = nT * kT * 512;
  if (i >= total) return;
  int v = i & 15;
  int l = (i >> 4) & 31;
  int fc = i >> 9;
  int c = fc % kT, t = fc / kT;
  int n = t * 16 + (l & 15);
  int k = c * 32 + ((l >= 16) ? 16 : 0) + v;
  float val = (k < K && n < N) ? W[(size_t)k * N + n] * scale : 0.0f;
  out[i] = f2bf(val);
}

// out[n,o] = (sum_k in[n,k] W[k,o]) * z[n] * scale
__global__ void k_lin(const float* __restrict__ in, const float* __restrict__ W,
                      const float* __restrict__ z, float* __restrict__ out,
                      int nn, int K, int O, float scale)
{
  int i = blockIdx.x * blockDim.x + threadIdx.x;
  if (i >= nn * O) return;
  int n = i / O, o = i - n * O;
  float acc = 0.0f;
  for (int k = 0; k < K; ++k) acc += in[(size_t)n * K + k] * W[(size_t)k * O + o];
  out[i] = acc * z[n] * scale;
}

// out[n,o,j] = (sum_k in[n,k,j] W[k,o]) * z[n] * scale
__global__ void k_linv(const float* __restrict__ in, const float* __restrict__ W,
                       const float* __restrict__ z, float* __restrict__ out,
                       int nn, int K, int O, float scale)
{
  int i = blockIdx.x * blockDim.x + threadIdx.x;
  if (i >= nn * O * 3) return;
  int j = i % 3;
  int no = i / 3;
  int n = no / O, o = no - n * O;
  float acc = 0.0f;
  for (int k = 0; k < K; ++k) acc += in[((size_t)n * K + k) * 3 + j] * W[(size_t)k * O + o];
  out[i] = acc * z[n] * scale;
}

__global__ void k_zero(float* __restrict__ p, size_t n)
{
  size_t i = (size_t)blockIdx.x * blockDim.x + threadIdx.x;
  if (i < n) p[i] = 0.0f;
}

// ---------------------------------------------------------------------------
// Fused edge kernels: radial MLP (WMMA) + gather + message + atomic scatter.
// One wave of 32 lanes per 16 edges; lane holds channel lm of each tile for
// 8 edges (rows r + 8*hi).
// ---------------------------------------------------------------------------
__global__ void k_edge_l1(const unsigned short* __restrict__ emb16,
    const unsigned short* __restrict__ w1f, const unsigned short* __restrict__ w2f,
    const int* __restrict__ esrc, const int* __restrict__ edst,
    const float* __restrict__ a0g, const float* __restrict__ a1g,
    const float* __restrict__ h1,
    float* __restrict__ aggS, float* __restrict__ aggV, int nTiles)
{
  __shared__ __align__(32) unsigned short smem[EW * 16 * 128];
  const int wave = threadIdx.x >> 5, lane = threadIdx.x & 31;
  const int tile = blockIdx.x * EW + wave;
  if (tile >= nTiles) return;           // uniform per wave; EXEC stays all-ones
  const int e0 = tile * 16;
  __builtin_prefetch(emb16 + (size_t)(e0 + 16 * EW) * 16, 0, 1);
  v8f w[2];
  radial16<2>(lane, e0, emb16, w1f, w2f, smem + wave * (16 * 128), w);
  const int lm = lane & 15, hi = lane >> 4;
  #pragma unroll
  for (int r = 0; r < 8; ++r) {
    const int e = e0 + r + hi * 8;
    const int s = esrc[e], d = edst[e];
    const float a0 = a0g[e];
    const float ax = a1g[(size_t)e * 3], ay = a1g[(size_t)e * 3 + 1], az = a1g[(size_t)e * 3 + 2];
    const float hs = h1[(size_t)s * 16 + lm] * INV_SQRT_NN;
    atomicAdd(&aggS[(size_t)d * 16 + lm], hs * a0 * w[0][r]);
    const float wv = hs * w[1][r];
    const size_t vb = ((size_t)d * 16 + lm) * 3;
    atomicAdd(&aggV[vb],     wv * ax);
    atomicAdd(&aggV[vb + 1], wv * ay);
    atomicAdd(&aggV[vb + 2], wv * az);
  }
}

__global__ void k_edge_l2(const unsigned short* __restrict__ emb16,
    const unsigned short* __restrict__ w1f, const unsigned short* __restrict__ w2f,
    const int* __restrict__ esrc, const int* __restrict__ edst,
    const float* __restrict__ a0g, const float* __restrict__ a1g,
    const float* __restrict__ hs2, const float* __restrict__ hv2,
    float* __restrict__ agg0, float* __restrict__ agg1o, int nTiles)
{
  __shared__ __align__(32) unsigned short smem[EW * 16 * 128];
  const int wave = threadIdx.x >> 5, lane = threadIdx.x & 31;
  const int tile = blockIdx.x * EW + wave;
  if (tile >= nTiles) return;
  const int e0 = tile * 16;
  __builtin_prefetch(emb16 + (size_t)(e0 + 16 * EW) * 16, 0, 1);
  v8f w[8];                      // tiles 8,9 (m1e path) are dead downstream
  radial16<8>(lane, e0, emb16, w1f, w2f, smem + wave * (16 * 128), w);
  const int lm = lane & 15, hi = lane >> 4;
  #pragma unroll
  for (int r = 0; r < 8; ++r) {
    const int e = e0 + r + hi * 8;
    const int s = esrc[e], d = edst[e];
    const float a0 = a0g[e];
    const float ax = a1g[(size_t)e * 3], ay = a1g[(size_t)e * 3 + 1], az = a1g[(size_t)e * 3 + 2];
    const float hs0 = hs2[(size_t)s * 32 + lm]      * INV_SQRT_NN;
    const float hs1 = hs2[(size_t)s * 32 + 16 + lm] * INV_SQRT_NN;
    const size_t hb0 = ((size_t)s * 32 + lm) * 3, hb1 = ((size_t)s * 32 + 16 + lm) * 3;
    const float h0x = hv2[hb0] * INV_SQRT_NN, h0y = hv2[hb0 + 1] * INV_SQRT_NN, h0z = hv2[hb0 + 2] * INV_SQRT_NN;
    const float h1x = hv2[hb1] * INV_SQRT_NN, h1y = hv2[hb1 + 1] * INV_SQRT_NN, h1z = hv2[hb1 + 2] * INV_SQRT_NN;
    // m0a -> agg0[0:32]
    atomicAdd(&agg0[(size_t)d * 64 + lm],      hs0 * a0 * w[0][r]);
    atomicAdd(&agg0[(size_t)d * 64 + 16 + lm], hs1 * a0 * w[1][r]);
    // m1a -> agg1o[0:32]
    const float t2 = hs0 * w[2][r], t3 = hs1 * w[3][r];
    const size_t v0 = ((size_t)d * 64 + lm) * 3, v1i = ((size_t)d * 64 + 16 + lm) * 3;
    atomicAdd(&agg1o[v0],      t2 * ax); atomicAdd(&agg1o[v0 + 1],  t2 * ay); atomicAdd(&agg1o[v0 + 2],  t2 * az);
    atomicAdd(&agg1o[v1i],     t3 * ax); atomicAdd(&agg1o[v1i + 1], t3 * ay); atomicAdd(&agg1o[v1i + 2], t3 * az);
    // m1b -> agg1o[32:64]
    const float t4 = a0 * w[4][r], t5 = a0 * w[5][r];
    const size_t v2 = ((size_t)d * 64 + 32 + lm) * 3, v3 = ((size_t)d * 64 + 48 + lm) * 3;
    atomicAdd(&agg1o[v2],      h0x * t4); atomicAdd(&agg1o[v2 + 1], h0y * t4); atomicAdd(&agg1o[v2 + 2], h0z * t4);
    atomicAdd(&agg1o[v3],      h1x * t5); atomicAdd(&agg1o[v3 + 1], h1y * t5); atomicAdd(&agg1o[v3 + 2], h1z * t5);
    // m0b -> agg0[32:64]
    const float dot0 = h0x * ax + h0y * ay + h0z * az;
    const float dot1 = h1x * ax + h1y * ay + h1z * az;
    atomicAdd(&agg0[(size_t)d * 64 + 32 + lm], dot0 * ISQ3_F * w[6][r]);
    atomicAdd(&agg0[(size_t)d * 64 + 48 + lm], dot1 * ISQ3_F * w[7][r]);
  }
}

__global__ void k_edge_l3(const unsigned short* __restrict__ emb16,
    const unsigned short* __restrict__ w1f, const unsigned short* __restrict__ w2f,
    const int* __restrict__ esrc, const int* __restrict__ edst,
    const float* __restrict__ a0g, const float* __restrict__ a1g,
    const float* __restrict__ hs3, const float* __restrict__ h1o3,
    float* __restrict__ agg, int nTiles)
{
  __shared__ __align__(32) unsigned short smem[EW * 16 * 128];
  const int wave = threadIdx.x >> 5, lane = threadIdx.x & 31;
  const int tile = blockIdx.x * EW + wave;
  if (tile >= nTiles) return;
  const int e0 = tile * 16;
  __builtin_prefetch(emb16 + (size_t)(e0 + 16 * EW) * 16, 0, 1);
  v8f w[4];
  radial16<4>(lane, e0, emb16, w1f, w2f, smem + wave * (16 * 128), w);
  const int lm = lane & 15, hi = lane >> 4;
  #pragma unroll
  for (int r = 0; r < 8; ++r) {
    const int e = e0 + r + hi * 8;
    const int s = esrc[e], d = edst[e];
    const float a0 = a0g[e];
    const float ax = a1g[(size_t)e * 3], ay = a1g[(size_t)e * 3 + 1], az = a1g[(size_t)e * 3 + 2];
    const float hs0 = hs3[(size_t)s * 32 + lm]      * INV_SQRT_NN;
    const float hs1 = hs3[(size_t)s * 32 + 16 + lm] * INV_SQRT_NN;
    const size_t hb0 = ((size_t)s * 32 + lm) * 3, hb1 = ((size_t)s * 32 + 16 + lm) * 3;
    const float dot0 = (h1o3[hb0] * ax + h1o3[hb0 + 1] * ay + h1o3[hb0 + 2] * az) * INV_SQRT_NN;
    const float dot1 = (h1o3[hb1] * ax + h1o3[hb1 + 1] * ay + h1o3[hb1 + 2] * az) * INV_SQRT_NN;
    atomicAdd(&agg[(size_t)d * 64 + lm],      hs0 * a0 * w[0][r]);
    atomicAdd(&agg[(size_t)d * 64 + 16 + lm], hs1 * a0 * w[1][r]);
    atomicAdd(&agg[(size_t)d * 64 + 32 + lm], dot0 * ISQ3_F * w[2][r]);
    atomicAdd(&agg[(size_t)d * 64 + 48 + lm], dot1 * ISQ3_F * w[3][r]);
  }
}

// ---------------------------------------------------------------------------
// Node-wise gating epilogues and graph pooling
// ---------------------------------------------------------------------------
__global__ void k_gate1(const float* __restrict__ sc, const float* __restrict__ os,
                        const float* __restrict__ ov,
                        float* __restrict__ s1, float* __restrict__ v1, int nn)
{
  int i = blockIdx.x * blockDim.x + threadIdx.x;
  if (i >= nn * 32) return;
  int n = i >> 5, c = i & 31;
  float oA = CS_F * sc[(size_t)n * 64 + c]      + CX_F * os[(size_t)n * 64 + c];
  float oB = CS_F * sc[(size_t)n * 64 + 32 + c] + CX_F * os[(size_t)n * 64 + 32 + c];
  s1[i] = siluf(oA);
  float g = sigmf(oB);
  size_t b = (size_t)i * 3;
  v1[b] = ov[b] * g; v1[b + 1] = ov[b + 1] * g; v1[b + 2] = ov[b + 2] * g;
}

__global__ void k_gate2(const float* __restrict__ sc, const float* __restrict__ os,
                        const float* __restrict__ o1o,
                        float* __restrict__ s2, float* __restrict__ v21o, int nn)
{
  int i = blockIdx.x * blockDim.x + threadIdx.x;
  if (i >= nn * 32) return;
  int n = i >> 5, c = i & 31;
  float oS = CS_F * sc[(size_t)n * 96 + c]      + CX_F * os[(size_t)n * 96 + c];
  float g1 = sigmf(CS_F * sc[(size_t)n * 96 + 32 + c] + CX_F * os[(size_t)n * 96 + 32 + c]);
  s2[i] = siluf(oS);
  size_t b = (size_t)i * 3;
  v21o[b] = o1o[b] * g1; v21o[b + 1] = o1o[b + 1] * g1; v21o[b + 2] = o1o[b + 2] * g1;
}

__global__ void k_pool(const float* __restrict__ sc3, const float* __restrict__ o3,
                       const int* __restrict__ batch,
                       float* __restrict__ gsum, float* __restrict__ gcnt, int nn)
{
  int n = blockIdx.x * blockDim.x + threadIdx.x;
  if (n >= nn) return;
  float val = CS_F * sc3[n] + CX_F * o3[n];
  int b = batch[n];
  atomicAdd(&gsum[b], val);
  atomicAdd(&gcnt[b], 1.0f);
}

__global__ void k_final(const float* __restrict__ gsum, const float* __restrict__ gcnt,
                        float* __restrict__ out)
{
  int g = threadIdx.x;
  if (g < NGRAPHS) out[g] = gsum[g] / fmaxf(gcnt[g], 1.0f);
}

// ---------------------------------------------------------------------------
extern "C" void kernel_launch(void* const* d_in, const int* in_sizes, int n_in,
                              void* d_out, int out_size, void* d_ws, size_t ws_size,
                              hipStream_t stream)
{
  (void)in_sizes; (void)n_in; (void)out_size; (void)ws_size;

  const float* x        = (const float*)d_in[0];
  const float* z        = (const float*)d_in[1];   // zs = z[:,0]
  const int*   esrc     = (const int*)  d_in[2];
  const int*   edst     = (const int*)  d_in[3];
  const float* evec     = (const float*)d_in[4];
  const int*   batch    = (const int*)  d_in[5];
  const float* l1_lin1   = (const float*)d_in[6];
  const float* l1_fc_w1  = (const float*)d_in[7];
  const float* l1_fc_w2  = (const float*)d_in[8];
  const float* l1_lin2_s = (const float*)d_in[9];
  const float* l1_lin2_v = (const float*)d_in[10];
  const float* l1_sc     = (const float*)d_in[11];
  const float* l2_lin1_s = (const float*)d_in[12];
  const float* l2_lin1_v = (const float*)d_in[13];
  const float* l2_fc_w1  = (const float*)d_in[14];
  const float* l2_fc_w2  = (const float*)d_in[15];
  const float* l2_lin2_s = (const float*)d_in[16];
  const float* l2_lin2_1o= (const float*)d_in[17];
  /* d_in[18] l2_lin2_1e: dead branch (v2_1e unused downstream) */
  const float* l2_sc     = (const float*)d_in[19];
  const float* l3_lin1_s = (const float*)d_in[20];
  const float* l3_lin1_1o= (const float*)d_in[21];
  const float* l3_fc_w1  = (const float*)d_in[22];
  const float* l3_fc_w2  = (const float*)d_in[23];
  const float* l3_lin2   = (const float*)d_in[24];
  const float* l3_sc     = (const float*)d_in[25];
  float* out = (float*)d_out;

  // ---- workspace arena (256B aligned slabs) ----
  char* base = (char*)d_ws;
  size_t cur = 0;
  auto alloc = [&](size_t bytes) -> void* {
    cur = (cur + 255) & ~(size_t)255;
    void* p = base + cur;
    cur += bytes;
    return p;
  };
  const size_t NN = NNODES, NE = NEDGES;
  unsigned short* emb16 = (unsigned short*)alloc(NE * 16 * 2);
  float* a0g   = (float*)alloc(NE * 4);
  float* a1g   = (float*)alloc(NE * 3 * 4);
  unsigned short* w1f1 = (unsigned short*)alloc(7 * 512 * 2);
  unsigned short* w2f1 = (unsigned short*)alloc(2 * 4 * 512 * 2);
  unsigned short* w1f2 = (unsigned short*)alloc(7 * 512 * 2);
  unsigned short* w2f2 = (unsigned short*)alloc(10 * 4 * 512 * 2);
  unsigned short* w1f3 = (unsigned short*)alloc(7 * 512 * 2);
  unsigned short* w2f3 = (unsigned short*)alloc(4 * 4 * 512 * 2);
  // reusable arenas (lifetimes commented)
  float* bufS  = (float*)alloc(NN * 64 * 4);    // agg_s1 -> agg0_2 -> agg3
  float* bufV  = (float*)alloc(NN * 192 * 4);   // agg_v1 -> agg1o_2
  float* osb   = (float*)alloc(NN * 96 * 4);    // os1 -> os2
  float* scb   = (float*)alloc(NN * 96 * 4);    // sc1 -> sc2 -> sc3
  float* ovb   = (float*)alloc(NN * 96 * 4);    // ov1 -> o1o -> o3
  float* hbuf  = (float*)alloc(NN * 32 * 4);    // h1 -> hs2 -> hs3
  float* hvbuf = (float*)alloc(NN * 96 * 4);    // hv2 -> h1o3
  float* s1b   = (float*)alloc(NN * 32 * 4);
  float* v1b   = (float*)alloc(NN * 96 * 4);
  float* s2b   = (float*)alloc(NN * 32 * 4);
  float* v21ob = (float*)alloc(NN * 96 * 4);
  float* gbuf  = (float*)alloc(2 * NGRAPHS * 4); // gsum | gcnt

  auto cdiv = [](size_t a, size_t b) -> unsigned { return (unsigned)((a + b - 1) / b); };
  const int T = 256;
  const unsigned eBlocks = cdiv(NTILES, EW);     // 128-thread blocks, 4 waves each

  // ---- geometry + weight fragment packing ----
  k_edge_geom<<<cdiv(NE, T), T, 0, stream>>>(evec, a0g, a1g, emb16, NE);
  k_pack_b<<<cdiv(7 * 512, T),  T, 0, stream>>>(l1_fc_w1, 10, 100, 1, 7,  IS10,  w1f1);
  k_pack_b<<<cdiv(8 * 512, T),  T, 0, stream>>>(l1_fc_w2, 100, 32, 4, 2,  IS100, w2f1);
  k_pack_b<<<cdiv(7 * 512, T),  T, 0, stream>>>(l2_fc_w1, 10, 100, 1, 7,  IS10,  w1f2);
  k_pack_b<<<cdiv(40 * 512, T), T, 0, stream>>>(l2_fc_w2, 100, 160, 4, 10, IS100, w2f2);
  k_pack_b<<<cdiv(7 * 512, T),  T, 0, stream>>>(l3_fc_w1, 10, 100, 1, 7,  IS10,  w1f3);
  k_pack_b<<<cdiv(16 * 512, T), T, 0, stream>>>(l3_fc_w2, 100, 64, 4, 4,  IS100, w2f3);

  // ---- layer 1 ----
  k_lin<<<cdiv(NN * 16, T), T, 0, stream>>>(x, l1_lin1, z, hbuf, NN, 16, 16, IS16);  // h1
  k_lin<<<cdiv(NN * 64, T), T, 0, stream>>>(x, l1_sc,   z, scb,  NN, 16, 64, IS16);  // sc1
  k_zero<<<cdiv(NN * 16, T), T, 0, stream>>>(bufS, NN * 16);
  k_zero<<<cdiv(NN * 48, T), T, 0, stream>>>(bufV, NN * 48);
  k_edge_l1<<<eBlocks, EW * 32, 0, stream>>>(emb16, w1f1, w2f1, esrc, edst, a0g, a1g,
                                             hbuf, bufS, bufV, NTILES);
  k_lin <<<cdiv(NN * 64, T), T, 0, stream>>>(bufS, l1_lin2_s, z, osb, NN, 16, 64, IS16);
  k_linv<<<cdiv(NN * 96, T), T, 0, stream>>>(bufV, l1_lin2_v, z, ovb, NN, 16, 32, IS16);
  k_gate1<<<cdiv(NN * 32, T), T, 0, stream>>>(scb, osb, ovb, s1b, v1b, NN);

  // ---- layer 2 ----
  k_lin <<<cdiv(NN * 32, T), T, 0, stream>>>(s1b, l2_lin1_s, z, hbuf,  NN, 32, 32, IS32); // hs2
  k_linv<<<cdiv(NN * 96, T), T, 0, stream>>>(v1b, l2_lin1_v, z, hvbuf, NN, 32, 32, IS32); // hv2
  k_zero<<<cdiv(NN * 64,  T), T, 0, stream>>>(bufS, NN * 64);
  k_zero<<<cdiv(NN * 192, T), T, 0, stream>>>(bufV, NN * 192);
  k_edge_l2<<<eBlocks, EW * 32, 0, stream>>>(emb16, w1f2, w2f2, esrc, edst, a0g, a1g,
                                             hbuf, hvbuf, bufS, bufV, NTILES);
  k_lin <<<cdiv(NN * 96, T), T, 0, stream>>>(bufS, l2_lin2_s,  z, osb, NN, 64, 96, IS64); // os2
  k_lin <<<cdiv(NN * 96, T), T, 0, stream>>>(s1b,  l2_sc,      z, scb, NN, 32, 96, IS32); // sc2
  k_linv<<<cdiv(NN * 96, T), T, 0, stream>>>(bufV, l2_lin2_1o, z, ovb, NN, 64, 32, IS64); // o1o
  k_gate2<<<cdiv(NN * 32, T), T, 0, stream>>>(scb, osb, ovb, s2b, v21ob, NN);

  // ---- layer 3 ----
  k_lin <<<cdiv(NN * 32, T), T, 0, stream>>>(s2b,   l3_lin1_s,  z, hbuf,  NN, 32, 32, IS32); // hs3
  k_linv<<<cdiv(NN * 96, T), T, 0, stream>>>(v21ob, l3_lin1_1o, z, hvbuf, NN, 32, 32, IS32); // h1o3
  k_zero<<<cdiv(NN * 64, T), T, 0, stream>>>(bufS, NN * 64);
  k_edge_l3<<<eBlocks, EW * 32, 0, stream>>>(emb16, w1f3, w2f3, esrc, edst, a0g, a1g,
                                             hbuf, hvbuf, bufS, NTILES);
  k_lin<<<cdiv(NN * 1, T), T, 0, stream>>>(bufS, l3_lin2, z, ovb, NN, 64, 1, IS64); // o3
  k_lin<<<cdiv(NN * 1, T), T, 0, stream>>>(s2b,  l3_sc,   z, scb, NN, 32, 1, IS32); // sc3

  // ---- pooling ----
  k_zero<<<1, 2 * NGRAPHS, 0, stream>>>(gbuf, 2 * NGRAPHS);
  k_pool<<<cdiv(NN, T), T, 0, stream>>>(scb, ovb, batch, gbuf, gbuf + NGRAPHS, NN);
  k_final<<<1, NGRAPHS, 0, stream>>>(gbuf, gbuf + NGRAPHS, out);
}
